// LRNN_20590073217055
// MI455X (gfx1250) — compile-verified, matrix operand order checked by reading, not run on hardware
//
#include <hip/hip_runtime.h>
#include <hip/hip_bf16.h>

// Problem dims (fixed by the reference)
#define T_DIM 512
#define B_DIM 64
#define NI    1024
#define NH    2048
#define NBLK  (NH / 16)          // 128 persistent blocks

// ---- CDNA5 WMMA types (wave32) ----
typedef __attribute__((ext_vector_type(16))) __bf16        v16bf;
typedef __attribute__((ext_vector_type(8)))  float         v8f;
typedef __attribute__((ext_vector_type(4)))  float         f32x4;
typedef __attribute__((ext_vector_type(4)))  unsigned int  u32x4;

// 16-bit A/B fragment: 16 bf16 = 8 dwords per lane.
// Element order per ISA layout: elems 0..7 = K off+0..7, elems 8..15 = K off+16..23
// (off = 0 for lanes 0..15, off = 8 for lanes 16..31).
union FragU {
    v16bf        v;
    unsigned int u[8];
    u32x4        q[2];
};

__device__ __forceinline__ unsigned int bf16_rne(float f) {
    unsigned int u = __float_as_uint(f);
    return (u + 0x7fffu + ((u >> 16) & 1u)) >> 16;
}
__device__ __forceinline__ unsigned int pack_bf16x2(float lo, float hi) {
    return (bf16_rne(lo) & 0xffffu) | (bf16_rne(hi) << 16);
}

// Load 16 f32 (two aligned groups of 8: p[0..7] and p[16..23]) and convert to a bf16 fragment.
__device__ __forceinline__ void load_frag_cvt_f32(const float* __restrict__ p, FragU& f) {
    f32x4 a0 = ((const f32x4*)p)[0];
    f32x4 a1 = ((const f32x4*)p)[1];
    f32x4 b0 = ((const f32x4*)(p + 16))[0];
    f32x4 b1 = ((const f32x4*)(p + 16))[1];
    f.u[0] = pack_bf16x2(a0.x, a0.y);
    f.u[1] = pack_bf16x2(a0.z, a0.w);
    f.u[2] = pack_bf16x2(a1.x, a1.y);
    f.u[3] = pack_bf16x2(a1.z, a1.w);
    f.u[4] = pack_bf16x2(b0.x, b0.y);
    f.u[5] = pack_bf16x2(b0.z, b0.w);
    f.u[6] = pack_bf16x2(b1.x, b1.y);
    f.u[7] = pack_bf16x2(b1.z, b1.w);
}

// Load a bf16 fragment directly (same two-group pattern), 16B-aligned b128 loads.
__device__ __forceinline__ void load_frag_bf16(const unsigned short* __restrict__ p, FragU& f) {
    f.q[0] = *(const u32x4*)(p);
    f.q[1] = *(const u32x4*)(p + 16);
}

// ---------------------------------------------------------------------------
// One-time f32 -> bf16 matrix conversion (8 elements / thread, b128 in/out).
// ---------------------------------------------------------------------------
__global__ void cvt_bf16_kernel(const float* __restrict__ src,
                                unsigned short* __restrict__ dst) {
    const size_t i = ((size_t)blockIdx.x * blockDim.x + threadIdx.x) * 8;
    f32x4 v0 = ((const f32x4*)(src + i))[0];
    f32x4 v1 = ((const f32x4*)(src + i))[1];
    u32x4 o;
    o.x = pack_bf16x2(v0.x, v0.y);
    o.y = pack_bf16x2(v0.z, v0.w);
    o.z = pack_bf16x2(v1.x, v1.y);
    o.w = pack_bf16x2(v1.z, v1.w);
    *(u32x4*)(dst + i) = o;
}

// ---------------------------------------------------------------------------
// init: a_bf16 = bf16(tanh(broadcast(Oh_X))); zero the 512 barrier counters.
// ---------------------------------------------------------------------------
__global__ void lrnn_init_kernel(const float* __restrict__ OhX,
                                 unsigned short* __restrict__ abf,
                                 unsigned int* __restrict__ ctr) {
    int idx = blockIdx.x * blockDim.x + threadIdx.x;   // 0 .. B*NH-1
    abf[idx] = (unsigned short)bf16_rne(tanhf(OhX[idx & (NH - 1)]));
    if (idx < T_DIM) ctr[idx] = 0u;
}

// ---------------------------------------------------------------------------
// Phase 1: z[tb, n] = sum_i x[tb, i] * U_w[n, i] + U_b[n], written into d_out.
// Block: 256 threads = 8 waves. Wave w: M-tile = blockIdx.x*8+w, 4 N-tiles at
// blockIdx.y*4. A fragment (x, cvt-on-load) reused across the 4 N-tiles;
// B fragments are pre-converted bf16 (pure b128 loads).
// Grid: (32768/16/8, 2048/16/4) = (256, 32).
// ---------------------------------------------------------------------------
__global__ void lrnn_zgemm_kernel(const float* __restrict__ x,
                                  const unsigned short* __restrict__ Uwbf,
                                  const float* __restrict__ Ub,
                                  float* __restrict__ z) {
    const int lane = threadIdx.x & 31;
    const int w    = threadIdx.x >> 5;
    const int mt   = blockIdx.x * 8 + w;          // 0..2047
    const int nt0  = blockIdx.y * 4;              // 0..124
    const int nl   = lane & 15;
    const int off  = (lane < 16) ? 0 : 8;

    const float* __restrict__ arow = x + (size_t)((mt << 4) + nl) * NI + off;
    const unsigned short* __restrict__ b0 = Uwbf + (size_t)((nt0 << 4) + nl) * NI + off;

    v8f zero = {0.f, 0.f, 0.f, 0.f, 0.f, 0.f, 0.f, 0.f};
    v8f acc[4];
#pragma unroll
    for (int j = 0; j < 4; ++j) acc[j] = zero;

    for (int k = 0; k < NI; k += 32) {
        FragU a;
        load_frag_cvt_f32(arow + k, a);
#pragma unroll
        for (int j = 0; j < 4; ++j) {
            FragU b;
            load_frag_bf16(b0 + (size_t)(j << 4) * NI + k, b);
            acc[j] = __builtin_amdgcn_wmma_f32_16x16x32_bf16(
                false, a.v, false, b.v, (short)0, acc[j], false, false);
        }
    }

    const int mbase = (lane >> 4) << 3;           // 0 or 8
#pragma unroll
    for (int j = 0; j < 4; ++j) {
        const int col = ((nt0 + j) << 4) + nl;
        const float bias = Ub[col];
#pragma unroll
        for (int r = 0; r < 8; ++r) {
            const int row = (mt << 4) + mbase + r;
            z[(size_t)row * NH + col] = acc[j][r] + bias;
        }
    }
}

// ---------------------------------------------------------------------------
// Persistent recurrence kernel: ONE launch runs all 512 timesteps.
// Grid: 128 blocks x 256 threads (8 waves). Block owns one N-tile (16 hidden
// cols) x full M=64. Each wave owns a fixed K=256 slice of W: its 8 W
// fragments (64 VGPRs) are loaded ONCE and stay register-resident for all
// 512 steps. h also stays in registers (fixed tile ownership). Per step:
// 32 WMMAs/wave against freshly-loaded activation fragments, LDS reduction,
// elementwise finalization, then a device-wide release/acquire barrier
// (one counter per step; global_inv on the acquire side so next step's
// activation loads can't hit stale WGP$ lines).
// ---------------------------------------------------------------------------
__global__ void __launch_bounds__(256, 1)
lrnn_persist_kernel(const unsigned short* __restrict__ Wbf,
                    const float* __restrict__ alpha_p,
                    const float* __restrict__ OhX,
                    unsigned short* __restrict__ abf0,
                    unsigned short* __restrict__ abf1,
                    unsigned int* __restrict__ ctr,
                    float* __restrict__ out) {
    __shared__ float red[8][4][256];              // 32 KB

    const int lane = threadIdx.x & 31;
    const int w    = threadIdx.x >> 5;
    const int nt   = blockIdx.x;                  // 0..127
    const int nl   = lane & 15;
    const int off  = (lane < 16) ? 0 : 8;
    const int k0   = w << 8;                      // this wave's K slice base

    // --- Register-resident W slice: 8 fragments, reused for all 512 steps ---
    const unsigned short* __restrict__ wp = Wbf + (size_t)((nt << 4) + nl) * NH + off + k0;
    FragU Wreg[8];
#pragma unroll
    for (int ks = 0; ks < 8; ++ks) load_frag_bf16(wp + ks * 32, Wreg[ks]);

    // --- Per-thread output-element state (fixed across steps) ---
    const int tid = threadIdx.x;
    const int m   = tid >> 4;
    const int n   = tid & 15;
    const int col = (nt << 4) + n;
    const float alpha = 1.f / (1.f + __expf(-alpha_p[col]));
    const float h0    = OhX[col];
    float hreg[4];
    size_t gidx[4];
#pragma unroll
    for (int j = 0; j < 4; ++j) {
        hreg[j] = h0;                             // h starts as broadcast(Oh_X)
        gidx[j] = (size_t)((j << 4) + m) * NH + col;
    }

    const unsigned short* __restrict__ abase0 = abf0 + (size_t)nl * NH + off;
    const unsigned short* __restrict__ abase1 = abf1 + (size_t)nl * NH + off;

    v8f zero = {0.f, 0.f, 0.f, 0.f, 0.f, 0.f, 0.f, 0.f};

#pragma clang loop unroll(disable)
    for (int t = 0; t < T_DIM; ++t) {
        const unsigned short* __restrict__ ap = (t & 1) ? abase1 : abase0;
        unsigned short* __restrict__ aw       = (t & 1) ? abf0 : abf1;

        v8f acc[4];
#pragma unroll
        for (int j = 0; j < 4; ++j) acc[j] = zero;

#pragma unroll
        for (int ks = 0; ks < 8; ++ks) {
            const int k = k0 + ks * 32;
#pragma unroll
            for (int j = 0; j < 4; ++j) {
                FragU a;
                load_frag_bf16(ap + (size_t)(j << 4) * NH + k, a);
                acc[j] = __builtin_amdgcn_wmma_f32_16x16x32_bf16(
                    false, a.v, false, Wreg[ks].v, (short)0, acc[j], false, false);
            }
        }

        // Wave partials -> LDS in (m,n) order.
        const int mbase = (lane >> 4) << 3;
#pragma unroll
        for (int j = 0; j < 4; ++j)
#pragma unroll
            for (int r = 0; r < 8; ++r)
                red[w][j][((mbase + r) << 4) + nl] = acc[j][r];
        __syncthreads();

        // 4 output elements per thread.
        float* zrow = out + (size_t)t * (B_DIM * NH);
#pragma unroll
        for (int j = 0; j < 4; ++j) {
            float s = 0.f;
#pragma unroll
            for (int ww = 0; ww < 8; ++ww) s += red[ww][j][tid];
            const float hn = alpha * hreg[j] + zrow[gidx[j]] + s;
            hreg[j] = hn;
            const float th = tanhf(hn);
            zrow[gidx[j]] = th;                   // out[t] overwrites z[t] in place
            aw[gidx[j]]   = (unsigned short)bf16_rne(th);
        }

        // ---- Device-wide step barrier (release -> arrive -> spin -> acquire) ----
        __builtin_amdgcn_fence(__ATOMIC_RELEASE, "agent");   // flush stores to device scope
        if (tid == 0) {
            __hip_atomic_fetch_add(&ctr[t], 1u, __ATOMIC_ACQ_REL, __HIP_MEMORY_SCOPE_AGENT);
            while (__hip_atomic_load(&ctr[t], __ATOMIC_ACQUIRE, __HIP_MEMORY_SCOPE_AGENT)
                   < (unsigned)NBLK)
                __builtin_amdgcn_s_sleep(1);
        }
        __builtin_amdgcn_fence(__ATOMIC_ACQUIRE, "agent");   // invalidate stale WGP$ lines
        __syncthreads();                                     // also guards LDS reuse
    }

    // h_final straight into the tail of d_out (no workspace h, no memcpy).
    float* hfin = out + (size_t)T_DIM * (B_DIM * NH);
#pragma unroll
    for (int j = 0; j < 4; ++j) hfin[gidx[j]] = hreg[j];
}

// ---------------------------------------------------------------------------
extern "C" void kernel_launch(void* const* d_in, const int* in_sizes, int n_in,
                              void* d_out, int out_size, void* d_ws, size_t ws_size,
                              hipStream_t stream) {
    const float* x       = (const float*)d_in[0];   // [T,B,NI]
    const float* U_w     = (const float*)d_in[1];   // [NH,NI]
    const float* U_b     = (const float*)d_in[2];   // [NH]
    const float* W_w     = (const float*)d_in[3];   // [NH,NH]
    const float* Oh_X    = (const float*)d_in[4];   // [1,NH]
    const float* alpha_p = (const float*)d_in[5];   // [NH]

    float* out = (float*)d_out;                     // [T,B,NH] out  ++  [B,NH] h_final

    // Workspace carve-out (~12.5 MB total):
    //   W_bf16  : NH*NH bf16 (8 MB)  — register/L2 resident
    //   Uw_bf16 : NH*NI bf16 (4 MB)
    //   abf0/1  : B*NH  bf16 x2 (256 KB each) — ping-pong tanh(h) matrix
    //   ctr     : T_DIM u32 (2 KB)  — per-step grid-barrier counters (zeroed each call)
    char* wsp = (char*)d_ws;
    unsigned short* Wbf  = (unsigned short*)wsp;    wsp += (size_t)NH * NH * 2;
    unsigned short* Uwbf = (unsigned short*)wsp;    wsp += (size_t)NH * NI * 2;
    unsigned short* abf0 = (unsigned short*)wsp;    wsp += (size_t)B_DIM * NH * 2;
    unsigned short* abf1 = (unsigned short*)wsp;    wsp += (size_t)B_DIM * NH * 2;
    unsigned int*   ctr  = (unsigned int*)wsp;

    // One-time weight conversions (8 elems/thread).
    cvt_bf16_kernel<<<(NH * NH) / (256 * 8), 256, 0, stream>>>(W_w, Wbf);
    cvt_bf16_kernel<<<(NH * NI) / (256 * 8), 256, 0, stream>>>(U_w, Uwbf);

    // a0 = bf16(tanh(broadcast(Oh_X))); zero barrier counters (fresh every call).
    lrnn_init_kernel<<<(B_DIM * NH) / 256, 256, 0, stream>>>(Oh_X, abf0, ctr);

    // z_all = x @ U_w^T + U_b, written directly into d_out's [T,B,NH] region.
    lrnn_zgemm_kernel<<<dim3(256, 32), 256, 0, stream>>>(x, Uwbf, U_b, out);

    // All 512 recurrence steps in ONE persistent kernel (device-wide barriers).
    lrnn_persist_kernel<<<NBLK, 256, 0, stream>>>(Wbf, alpha_p, Oh_X,
                                                  abf0, abf1, ctr, out);
}
